// Net_455266533916
// MI455X (gfx1250) — matrix operand order repaired and checked by reading, compiled, not run on hardware
//
#include <hip/hip_runtime.h>
#include <math.h>

typedef __attribute__((ext_vector_type(16))) _Float16 v16h;
typedef __attribute__((ext_vector_type(8)))  float    v8f;

#define NB 4          // batch size (fixed by reference)
#define C1 64         // 2*FILTERS
#define C2 32         // FILTERS

// ---------------- utility ----------------
__global__ void k_zero_i32(int* p, int n) {
  int i = blockIdx.x * blockDim.x + threadIdx.x;
  if (i < n) p[i] = 0;
}

// deg[dst] += 1 per edge (self-loop +1 folded into k_dinv)
__global__ void k_deg(const int* __restrict__ ei, int E, int* __restrict__ deg) {
  int e = blockIdx.x * blockDim.x + threadIdx.x;
  if (e < E) atomicAdd(&deg[ei[E + e]], 1);
}

__global__ void k_dinv(const int* __restrict__ deg, float* __restrict__ dinv, int N) {
  int n = blockIdx.x * blockDim.x + threadIdx.x;
  if (n < N) dinv[n] = rsqrtf((float)deg[n] + 1.0f);   // +1 = self loop; always > 0
}

// ---------------- layer 1 (scalar propagation, Cin==1) ----------------
// s1[b,n] = dinv[n]^2 * x[b,n]   (self-loop term; also serves as the zero-init)
__global__ void k_s1_init(const float* __restrict__ x, const float* __restrict__ dinv,
                          float* __restrict__ s1, int N) {
  int i = blockIdx.x * blockDim.x + threadIdx.x;
  if (i < NB * N) {
    float w = dinv[i % N];
    s1[i] = w * w * x[i];
  }
}

__global__ void k_s1_edges(const int* __restrict__ ei, const float* __restrict__ x,
                           const float* __restrict__ dinv, float* __restrict__ s1,
                           int E, int N) {
  int e = blockIdx.x * blockDim.x + threadIdx.x;
  if (e >= E) return;
  int s = ei[e], d = ei[E + e];
  float nrm = dinv[s] * dinv[d];
  #pragma unroll
  for (int b = 0; b < NB; ++b)
    atomicAdd(&s1[b * N + d], nrm * x[b * N + s]);
}

// ---------------- layer 2 GEMM via WMMA (hi/lo split f16) ----------------
// y2pre[row, 0:32] = relu(W1[:]*s1[row] + b1[:]) @ W2[64,32]
// One wave32 per 16-row tile; K=64 as two 16x16x32 chunks x 3 split terms.
// W2 is pre-split (hi/lo f16) into LDS in the exact WMMA B-fragment register
// layout [hl][ct][ko][lane][j], so each fragment is two aligned ds_load_b128.
__global__ void k_mm2_wmma(const float* __restrict__ s1,
                           const float* __restrict__ W1g,
                           const float* __restrict__ B1g,
                           const float* __restrict__ W2g,
                           float* __restrict__ y2pre, int BN) {
  __shared__ __align__(32) _Float16 sBfrag[2][2][2][32][16]; // [hi0/lo1][ct][ko][lane][j]
  __shared__ float sW1[C1];
  __shared__ float sB1[C1];

  int tid = threadIdx.x;
  // Fill B fragments: contiguous k order, same as A -> half j of lane (g,r) holds
  // K = 32*ko + 16*g + j, N(col) = 16*ct + r.
  for (int idx = tid; idx < 4096; idx += 256) {
    int j    = idx & 15;
    int ln   = (idx >> 4) & 31;
    int ko   = (idx >> 9) & 1;
    int ct   = (idx >> 10) & 1;
    int hl   = (idx >> 11) & 1;
    int gg   = ln >> 4;
    int rr   = ln & 15;
    int k    = (ko << 5) + (gg << 4) + j;
    int col  = (ct << 4) + rr;
    float w  = W2g[k * C2 + col];
    _Float16 hi = (_Float16)w;
    sBfrag[hl][ct][ko][ln][j] = hl ? (_Float16)(w - (float)hi) : hi;
  }
  if (tid < C1) { sW1[tid] = W1g[tid]; sB1[tid] = B1g[tid]; }
  __syncthreads();

  int wave = tid >> 5;
  int lane = tid & 31;
  int r = lane & 15;     // row within tile (A/M), column (B,C/D N)
  int g = lane >> 4;     // lane-half selector
  int tile = blockIdx.x * 8 + wave;
  if (tile * 16 >= BN) return;              // wave-uniform exit: EXEC stays full for WMMA

  int rowA = tile * 16 + r;
  float s = (rowA < BN) ? s1[rowA] : 0.0f;

  // Build A hi/lo for K-chunks ko=0 (ahi0/alo0) and ko=32 (ahi1/alo1).
  // A uses contiguous k order (same as B): half j of lane (g,r) holds K = 16*g + j.
  v16h ahi0, alo0, ahi1, alo1;
  #pragma unroll
  for (int j = 0; j < 16; ++j) {
    int cb = (g << 4) + j;   // in [0,32)
    float v0 = fmaf(sW1[cb], s, sB1[cb]);
    v0 = v0 > 0.0f ? v0 : 0.0f;                      // ReLU of layer-1 output
    _Float16 h0 = (_Float16)v0;
    ahi0[j] = h0;
    alo0[j] = (_Float16)(v0 - (float)h0);

    int c1 = 32 + cb;
    float v1 = fmaf(sW1[c1], s, sB1[c1]);
    v1 = v1 > 0.0f ? v1 : 0.0f;
    _Float16 h1 = (_Float16)v1;
    ahi1[j] = h1;
    alo1[j] = (_Float16)(v1 - (float)h1);
  }

  bool fullTile = (tile * 16 + 16 <= BN);    // wave-uniform

  #pragma unroll
  for (int ct = 0; ct < 2; ++ct) {                   // two 16-column tiles of C2=32
    int d0 = ct << 4;
    v8f acc = {};
    #pragma unroll
    for (int ko = 0; ko < 2; ++ko) {                 // K chunks 0..31, 32..63
      v16h bhi = *(const v16h*)(&sBfrag[0][ct][ko][lane][0]);
      v16h blo = *(const v16h*)(&sBfrag[1][ct][ko][lane][0]);
      const v16h& ahi = ko ? ahi1 : ahi0;
      const v16h& alo = ko ? alo1 : alo0;
      acc = __builtin_amdgcn_wmma_f32_16x16x32_f16(false, ahi, false, bhi, (short)0, acc, false, false);
      acc = __builtin_amdgcn_wmma_f32_16x16x32_f16(false, ahi, false, blo, (short)0, acc, false, false);
      acc = __builtin_amdgcn_wmma_f32_16x16x32_f16(false, alo, false, bhi, (short)0, acc, false, false);
    }
    // C/D layout: VGPR v -> row M = v + 8*g, column N = r.
    if (fullTile) {
      // straight stores: base + v*128B immediate offsets
      float* op = y2pre + ((size_t)(tile * 16 + (g << 3))) * C2 + d0 + r;
      #pragma unroll
      for (int v = 0; v < 8; ++v) op[(size_t)v * C2] = acc[v];
    } else {
      #pragma unroll
      for (int v = 0; v < 8; ++v) {
        int row = tile * 16 + v + (g << 3);
        if (row < BN) y2pre[(size_t)row * C2 + d0 + r] = acc[v];
      }
    }
  }
}

// ---------------- layer 2 propagation ----------------
// y2agg = dinv^2 * y2pre  (self-loop; doubles as zero-init)
__global__ void k_y2_selfinit(const float* __restrict__ y2pre, const float* __restrict__ dinv,
                              float* __restrict__ y2agg, int N) {
  int i = blockIdx.x * blockDim.x + threadIdx.x;    // over NB*N nodes
  if (i >= NB * N) return;
  float w = dinv[i % N]; w = w * w;
  const float4* sp = (const float4*)(y2pre + (size_t)i * C2);
  float4* dp = (float4*)(y2agg + (size_t)i * C2);
  #pragma unroll
  for (int q = 0; q < 8; ++q) {
    float4 v = sp[q];
    dp[q] = make_float4(w * v.x, w * v.y, w * v.z, w * v.w);
  }
}

// one thread per (edge, batch): gather 32ch from src, atomic-scatter to dst
__global__ void k_y2_edges(const int* __restrict__ ei, const float* __restrict__ dinv,
                           const float* __restrict__ y2pre, float* __restrict__ y2agg,
                           int E, int N) {
  int gid = blockIdx.x * blockDim.x + threadIdx.x;
  if (gid >= NB * E) return;
  int e = gid >> 2;
  int b = gid & 3;
  int s = ei[e], d = ei[E + e];
  float nrm = dinv[s] * dinv[d];
  const float4* sp = (const float4*)(y2pre + ((size_t)b * N + s) * C2);
  float* dp = y2agg + ((size_t)b * N + d) * C2;
  #pragma unroll
  for (int q = 0; q < 8; ++q) {
    float4 v = sp[q];
    atomicAdd(dp + 4 * q + 0, nrm * v.x);
    atomicAdd(dp + 4 * q + 1, nrm * v.y);
    atomicAdd(dp + 4 * q + 2, nrm * v.z);
    atomicAdd(dp + 4 * q + 3, nrm * v.w);
  }
}

// ---------------- layer 3 (scalar) ----------------
// h3[b,n] = relu(y2agg + b2) . W3
__global__ void k_h3(const float* __restrict__ y2agg, const float* __restrict__ b2,
                     const float* __restrict__ W3, float* __restrict__ h3, int N) {
  int i = blockIdx.x * blockDim.x + threadIdx.x;
  if (i >= NB * N) return;
  const float* p = y2agg + (size_t)i * C2;
  float acc = 0.0f;
  #pragma unroll
  for (int d = 0; d < C2; ++d) {
    float v = p[d] + b2[d];
    acc = fmaf(fmaxf(v, 0.0f), W3[d], acc);
  }
  h3[i] = acc;
}

__global__ void k_out_init(const float* __restrict__ h3, const float* __restrict__ dinv,
                           float* __restrict__ out, int N) {
  int i = blockIdx.x * blockDim.x + threadIdx.x;
  if (i >= NB * N) return;
  float w = dinv[i % N];
  out[i] = w * w * h3[i];
}

__global__ void k_out_edges(const int* __restrict__ ei, const float* __restrict__ dinv,
                            const float* __restrict__ h3, float* __restrict__ out,
                            int E, int N) {
  int e = blockIdx.x * blockDim.x + threadIdx.x;
  if (e >= E) return;
  int s = ei[e], d = ei[E + e];
  float nrm = dinv[s] * dinv[d];
  #pragma unroll
  for (int b = 0; b < NB; ++b)
    atomicAdd(&out[b * N + d], nrm * h3[b * N + s]);
}

__global__ void k_final(float* __restrict__ out, const float* __restrict__ b3, int BN) {
  int i = blockIdx.x * blockDim.x + threadIdx.x;
  if (i >= BN) return;
  float t = out[i] + b3[0];
  out[i] = 1.0f / (1.0f + expf(-t));
}

// ---------------- launcher ----------------
extern "C" void kernel_launch(void* const* d_in, const int* in_sizes, int n_in,
                              void* d_out, int out_size, void* d_ws, size_t ws_size,
                              hipStream_t stream) {
  const float* x   = (const float*)d_in[0];      // [B,N,1]
  const int*   ei  = (const int*)d_in[1];        // [2,E]
  const float* W1  = (const float*)d_in[2];      // [1,64]
  const float* b1  = (const float*)d_in[3];      // [64]
  const float* W2  = (const float*)d_in[4];      // [64,32]
  const float* b2  = (const float*)d_in[5];      // [32]
  const float* W3  = (const float*)d_in[6];      // [32,1]
  const float* b3  = (const float*)d_in[7];      // [1]
  float* out = (float*)d_out;

  const int N  = in_sizes[0] / NB;
  const int E  = in_sizes[1] / 2;
  const int BN = NB * N;

  // workspace carve-up (256B aligned)
  char* ws = (char*)d_ws;
  size_t off = 0;
  auto take = [&](size_t bytes) -> char* {
    char* p = ws + off;
    off = (off + bytes + 255) & ~(size_t)255;
    return p;
  };
  int*   degI  = (int*)  take((size_t)N * 4);
  float* dinv  = (float*)take((size_t)N * 4);
  float* s1    = (float*)take((size_t)BN * 4);
  float* h3    = (float*)take((size_t)BN * 4);
  float* y2pre = (float*)take((size_t)BN * C2 * 4);
  float* y2agg = (float*)take((size_t)BN * C2 * 4);
  (void)ws_size; (void)n_in; (void)out_size;

  const int T = 256;
  // normalization
  k_zero_i32 <<<(N + T - 1) / T, T, 0, stream>>>(degI, N);
  k_deg      <<<(E + T - 1) / T, T, 0, stream>>>(ei, E, degI);
  k_dinv     <<<(N + T - 1) / T, T, 0, stream>>>(degI, dinv, N);
  // layer 1 (scalar propagation)
  k_s1_init  <<<(BN + T - 1) / T, T, 0, stream>>>(x, dinv, s1, N);
  k_s1_edges <<<(E + T - 1) / T, T, 0, stream>>>(ei, x, dinv, s1, E, N);
  // layer 2 GEMM (WMMA) + vector propagation
  int tiles = (BN + 15) / 16;
  k_mm2_wmma   <<<(tiles + 7) / 8, 256, 0, stream>>>(s1, W1, b1, W2, y2pre, BN);
  k_y2_selfinit<<<(BN + T - 1) / T, T, 0, stream>>>(y2pre, dinv, y2agg, N);
  k_y2_edges   <<<(NB * E + T - 1) / T, T, 0, stream>>>(ei, dinv, y2pre, y2agg, E, N);
  // layer 3 (scalar propagation) + sigmoid
  k_h3       <<<(BN + T - 1) / T, T, 0, stream>>>(y2agg, b2, W3, h3, N);
  k_out_init <<<(BN + T - 1) / T, T, 0, stream>>>(h3, dinv, out, N);
  k_out_edges<<<(E + T - 1) / T, T, 0, stream>>>(ei, dinv, h3, out, E, N);
  k_final    <<<(BN + T - 1) / T, T, 0, stream>>>(out, b3, BN);
}
